// LatentEwaldSum_14723147890762
// MI455X (gfx1250) — matrix operand order repaired and checked by reading, not compiled, not run on hardware
//
#include <hip/hip_runtime.h>

// Latent Ewald reciprocal sum for MI455X (gfx1250, wave32, WMMA).
// G=32 graphs, NA=1024 atoms, C=4 channels, K=2196 k-vectors (nmax=6).

#define G_      32
#define NA_     1024
#define C_      4
#define K_      2196
#define NT_     138            // ceil(K_/16) k-tiles of 16
#define KP_     (NT_ * 16)     // 2208: K padded to tile boundary (zeros)
#define ORIGIN_ 1098           // flat index of (0,0,0) in the 13^3 grid

typedef __attribute__((ext_vector_type(16))) _Float16 v16h;
typedef __attribute__((ext_vector_type(8)))  float    v8f;

#define TWO_PI_F 6.28318530717958647692f
#define PI_F     3.14159265358979323846f

// ---------------------------------------------------------------------------
// Kernel 1: per (g,k) reciprocal vectors (in "turns": kvec/2pi), weights, vol.
// Pads k in [K_, KP_) with zeros so the main kernel is branch-free.
// ---------------------------------------------------------------------------
__global__ __launch_bounds__(128)
void lew_setup(const float* __restrict__ cell,   // [G,3,3]
               float* __restrict__ kvt,          // [G,KP,3] (turn units)
               float* __restrict__ wBuf,         // [G,KP]
               float* __restrict__ vol)          // [G]
{
    const int g = blockIdx.y;
    const int k = blockIdx.x * blockDim.x + threadIdx.x;
    if (k >= KP_) return;

    float kx = 0.0f, ky = 0.0f, kz = 0.0f, wv = 0.0f;

    const float* c = cell + g * 9;
    const float c00 = c[0], c01 = c[1], c02 = c[2];
    const float c10 = c[3], c11 = c[4], c12 = c[5];
    const float c20 = c[6], c21 = c[7], c22 = c[8];

    const float det = c00 * (c11 * c22 - c12 * c21)
                    - c01 * (c10 * c22 - c12 * c20)
                    + c02 * (c10 * c21 - c11 * c20);

    if (k < K_) {
        const float id = 1.0f / det;
        // inv(cell)
        const float i00 = (c11 * c22 - c12 * c21) * id;
        const float i01 = (c02 * c21 - c01 * c22) * id;
        const float i02 = (c01 * c12 - c02 * c11) * id;
        const float i10 = (c12 * c20 - c10 * c22) * id;
        const float i11 = (c00 * c22 - c02 * c20) * id;
        const float i12 = (c02 * c10 - c00 * c12) * id;
        const float i20 = (c10 * c21 - c11 * c20) * id;
        const float i21 = (c01 * c20 - c00 * c21) * id;
        const float i22 = (c00 * c11 - c01 * c10) * id;

        // integer triple (iz fastest: meshgrid('ij').reshape(-1,3), origin cut)
        const int full = (k < ORIGIN_) ? k : k + 1;
        const float n2 = (float)(full % 13 - 6);
        const float n1 = (float)((full / 13) % 13 - 6);
        const float n0 = (float)(full / 169 - 6);

        // kvec_d / 2pi = sum_j n_j * inv[d][j]   (recip = 2pi * inv(cell)^T)
        kx = n0 * i00 + n1 * i01 + n2 * i02;
        ky = n0 * i10 + n1 * i11 + n2 * i12;
        kz = n0 * i20 + n1 * i21 + n2 * i22;

        const float k2 = (TWO_PI_F * TWO_PI_F) * (kx * kx + ky * ky + kz * kz);
        wv = 4.0f * PI_F * __expf(-0.5f * k2) / fmaxf(k2, 1e-12f);
    }

    float* kv = kvt + ((size_t)g * KP_ + k) * 3;
    kv[0] = kx; kv[1] = ky; kv[2] = kz;
    wBuf[(size_t)g * KP_ + k] = wv;
    if (k == 0) vol[g] = fabsf(det);
}

// ---------------------------------------------------------------------------
// Kernel 2: main WMMA contraction. 4 waves/block, one 16-k tile per wave.
//   D_re(16x16) += cosTile(16x32 f16) x qTile(32x16 f16)   [cols 0..3 live]
//   D_im(16x16) += sinTile(16x32 f16) x qTile(32x16 f16)
// Branch-free inner loop: SoA positions + 16-col zero-padded transposed q
// give aligned LDS vector loads that match the WMMA per-lane layouts exactly.
// ---------------------------------------------------------------------------
__global__ __launch_bounds__(128)
void lew_main(const float* __restrict__ q,     // [G*NA, C]
              const float* __restrict__ pos,   // [G*NA, 3]
              const float* __restrict__ kvt,   // [G,KP,3] (turns)
              const float* __restrict__ wBuf,  // [G,KP]
              float* __restrict__ ePart)       // [G, NT]
{
    __shared__ float    sPx[NA_], sPy[NA_], sPz[NA_];  // 12 KB (SoA)
    __shared__ _Float16 sQT[16 * NA_];                 // 32 KB, [col][atom]

    const int g   = blockIdx.y;
    const int tid = threadIdx.x;

    // Stage the whole graph once (inputs are L2-resident).
    for (int i = tid; i < NA_; i += 128) {
        const float* p = pos + ((size_t)g * NA_ + i) * 3;
        sPx[i] = p[0]; sPy[i] = p[1]; sPz[i] = p[2];
    }
    for (int i = tid; i < 16 * NA_; i += 128) {
        const int cidx = i >> 10;          // column 0..15
        const int n    = i & (NA_ - 1);    // atom
        const float val = q[((size_t)g * NA_ + n) * C_ + (cidx & 3)];
        sQT[i] = (cidx < C_) ? (_Float16)val : (_Float16)0.0f;
    }
    __syncthreads();

    const int tile = blockIdx.x * 4 + (tid >> 5);
    if (tile >= NT_) return;

    const int lane = tid & 31;
    const int hi   = lane >> 4;      // lane half
    const int mrow = lane & 15;      // A row M == B/C/D column N
    const int kt0  = tile * 16;

    // Warm L2/L0 for the epilogue weights (global_prefetch_b8).
    __builtin_prefetch(&wBuf[(size_t)g * KP_ + kt0], 0, 1);

    // k-vector (turn units) for this lane's A row (zero-padded tail).
    const float* kv = kvt + ((size_t)g * KP_ + kt0 + mrow) * 3;
    const float kx = kv[0], ky = kv[1], kz = kv[2];

    v8f accRe = {}; v8f accIm = {};

#pragma unroll 2
    for (int n0 = 0; n0 < NA_; n0 += 32) {
        // A atoms come in two runs of 8 consecutive: n0+hi*8, n0+16+hi*8.
        const int a0 = n0 + hi * 8;
        const int a1 = n0 + 16 + hi * 8;
        const v8f px0 = *(const v8f*)&sPx[a0], px1 = *(const v8f*)&sPx[a1];
        const v8f py0 = *(const v8f*)&sPy[a0], py1 = *(const v8f*)&sPy[a1];
        const v8f pz0 = *(const v8f*)&sPz[a0], pz1 = *(const v8f*)&sPz[a1];

        v16h aCos, aSin;
#pragma unroll
        for (int j = 0; j < 8; ++j) {
            // phase in turns: v_cos_f32/v_sin_f32 take revolutions
            const float t0 = fmaf(kx, px0[j], fmaf(ky, py0[j], kz * pz0[j]));
            const float t1 = fmaf(kx, px1[j], fmaf(ky, py1[j], kz * pz1[j]));
            aCos[j]     = (_Float16)__builtin_amdgcn_cosf(t0);
            aSin[j]     = (_Float16)__builtin_amdgcn_sinf(t0);
            aCos[8 + j] = (_Float16)__builtin_amdgcn_cosf(t1);
            aSin[8 + j] = (_Float16)__builtin_amdgcn_sinf(t1);
        }

        // B fragment: two runs of 8 f16 (16B aligned) -> 2x ds_load_b128.
        // elements 0..7: atoms n0+hi*8+e; elements 8..15: atoms n0+16+hi*8+(e-8); column mrow.
        typedef __attribute__((ext_vector_type(8))) _Float16 v8h_t;
        const v8h_t bq0 = *(const v8h_t*)&sQT[(size_t)mrow * NA_ + a0];
        const v8h_t bq1 = *(const v8h_t*)&sQT[(size_t)mrow * NA_ + a1];
        const v16h bQ = __builtin_shufflevector(bq0, bq1, 0,1,2,3,4,5,6,7,8,9,10,11,12,13,14,15);

        accRe = __builtin_amdgcn_wmma_f32_16x16x32_f16(
            false, aCos, false, bQ, (short)0, accRe, false, false);
        accIm = __builtin_amdgcn_wmma_f32_16x16x32_f16(
            false, aSin, false, bQ, (short)0, accIm, false, false);
    }

    // D layout (f32 16x16): VGPR r holds M = r + hi*8, N = mrow.
    // Columns N>=4 are exactly zero (B cols were zero), so sum everything.
    // Weights for this lane's 8 rows are consecutive: one aligned v8f load.
    const v8f wv = *(const v8f*)&wBuf[(size_t)g * KP_ + kt0 + hi * 8];
    float e = 0.0f;
#pragma unroll
    for (int r = 0; r < 8; ++r)
        e = fmaf(wv[r], accRe[r] * accRe[r] + accIm[r] * accIm[r], e);

#pragma unroll
    for (int s = 16; s > 0; s >>= 1) e += __shfl_xor(e, s, 32);
    if (lane == 0) ePart[(size_t)g * NT_ + tile] = e;
}

// ---------------------------------------------------------------------------
// Kernel 3: deterministic per-graph reduction, divide by 2V.
// ---------------------------------------------------------------------------
__global__ __launch_bounds__(128)
void lew_reduce(const float* __restrict__ ePart,  // [G, NT]
                const float* __restrict__ vol,    // [G]
                float* __restrict__ out)          // [G]
{
    __shared__ float red[128];
    const int g = blockIdx.x, tid = threadIdx.x;
    float s = 0.0f;
    for (int t = tid; t < NT_; t += 128) s += ePart[(size_t)g * NT_ + t];
    red[tid] = s;
    __syncthreads();
#pragma unroll
    for (int off = 64; off > 0; off >>= 1) {
        if (tid < off) red[tid] += red[tid + off];
        __syncthreads();
    }
    if (tid == 0) out[g] = red[0] / (2.0f * vol[g]);
}

// ---------------------------------------------------------------------------
extern "C" void kernel_launch(void* const* d_in, const int* in_sizes, int n_in,
                              void* d_out, int out_size, void* d_ws, size_t ws_size,
                              hipStream_t stream) {
    (void)in_sizes; (void)n_in; (void)out_size; (void)ws_size;
    const float* q    = (const float*)d_in[0];  // [G*NA, C]
    const float* pos  = (const float*)d_in[1];  // [G*NA, 3]
    const float* cell = (const float*)d_in[2];  // [G, 3, 3]
    // d_in[3] = batch (int), unused: equal-sized sorted segments.
    float* out = (float*)d_out;                 // [G]

    // Workspace layout (floats): kvt | w | vol | ePart  (~1.15 MB)
    float* kvt   = (float*)d_ws;
    float* wBuf  = kvt + (size_t)G_ * KP_ * 3;
    float* vol   = wBuf + (size_t)G_ * KP_;
    float* ePart = vol + G_;

    lew_setup<<<dim3((KP_ + 127) / 128, G_), 128, 0, stream>>>(cell, kvt, wBuf, vol);
    lew_main<<<dim3((NT_ + 3) / 4, G_), 128, 0, stream>>>(q, pos, kvt, wBuf, ePart);
    lew_reduce<<<G_, 128, 0, stream>>>(ePart, vol, out);
}
